// InnerSelfAttention_541165879742
// MI455X (gfx1250) — compile-verified, matrix operand order checked by reading, not run on hardware
//
#include <hip/hip_runtime.h>
#include <hip/hip_bf16.h>

// ---------------------------------------------------------------------------
// Sliding-window causal self-attention for MI455X (gfx1250, wave32, WMMA).
// All matmuls (QKV proj, Q*K^T, P*V, out proj) use v_wmma_f32_16x16x32_bf16.
// GEMMs stage weights through LDS with double-buffered async copies
// (global_load_async_to_lds_b128, ASYNCcnt-tracked).
// ---------------------------------------------------------------------------

typedef __bf16 v16bf __attribute__((ext_vector_type(16)));
typedef __bf16 v8bf  __attribute__((ext_vector_type(8)));
typedef __bf16 v4bf  __attribute__((ext_vector_type(4)));
typedef float  v8f   __attribute__((ext_vector_type(8)));
typedef int    v4i_vs __attribute__((vector_size(16)));   // matches builtin sig

namespace {
constexpr int kB  = 2;
constexpr int kS  = 2048;
constexpr int kD  = 1024;
constexpr int kH  = 16;
constexpr int kHD = 64;
constexpr int kW  = 256;
constexpr int kM  = kB * kS;        // 4096 rows
constexpr int KCH = 64;             // K-chunk staged in LDS per GEMM step
}

#if __has_builtin(__builtin_amdgcn_global_load_async_to_lds_b128)
#define ASYNC_LDS_COPY 1
#else
#define ASYNC_LDS_COPY 0
#endif

// Copy 16 bytes global -> LDS. Async (ASYNCcnt-tracked) when available.
__device__ __forceinline__ void copy16_g2l(const __bf16* g, __bf16* l) {
#if ASYNC_LDS_COPY
  auto gp = (__attribute__((address_space(1))) v4i_vs*)(uintptr_t)g;
  auto lp = (__attribute__((address_space(3))) v4i_vs*)(uint32_t)(uintptr_t)l;
  __builtin_amdgcn_global_load_async_to_lds_b128(gp, lp, 0, 0);
#else
  *reinterpret_cast<v8bf*>(l) = *reinterpret_cast<const v8bf*>(g);
#endif
}

__device__ __forceinline__ void wait_async_copies() {
#if ASYNC_LDS_COPY
#if __has_builtin(__builtin_amdgcn_s_wait_asynccnt)
  __builtin_amdgcn_s_wait_asynccnt(0);
#else
  asm volatile("s_wait_asynccnt 0x0" ::: "memory");
#endif
#endif
}

// Load a 16x32 bf16 WMMA fragment (A or B) from a row-major source.
// rowp points at the start of this lane's row; K-slot pattern per ISA:
//   lanes 0-15  : K = {k0+0..7,  k0+16..23}
//   lanes 16-31 : K = {k0+8..15, k0+24..31}
__device__ __forceinline__ v16bf ldfrag(const __bf16* rowp, int k0, int half) {
  const __bf16* p = rowp + k0 + half * 8;
  union U { v16bf v; v8bf h[2]; } u;
  u.h[0] = *reinterpret_cast<const v8bf*>(p);
  u.h[1] = *reinterpret_cast<const v8bf*>(p + 16);
  return u.v;
}

__device__ __forceinline__ v8f wmma_bf16(v16bf a, v16bf b, v8f c) {
  return __builtin_amdgcn_wmma_f32_16x16x32_bf16(
      /*neg_a=*/false, a, /*neg_b=*/false, b,
      /*c_mod=*/(short)0, c, /*reuse_a=*/false, /*reuse_b=*/false);
}

// ---------------------------------------------------------------------------
__global__ void f32_to_bf16_kernel(const float* __restrict__ in,
                                   __bf16* __restrict__ out, int n4) {
  int i = blockIdx.x * blockDim.x + threadIdx.x;
  if (i < n4) {
    const float4 f = reinterpret_cast<const float4*>(in)[i];
    v4bf o;
    o[0] = (__bf16)f.x; o[1] = (__bf16)f.y;
    o[2] = (__bf16)f.z; o[3] = (__bf16)f.w;
    reinterpret_cast<v4bf*>(out)[i] = o;
  }
}

// ---------------------------------------------------------------------------
// C = A[4096xK] * W^T, W is [1024 x K] row-major (row n = output feature n).
// Block (8 waves) computes a 128x64 tile: all waves share one 64-wide weight
// strip staged in LDS (double-buffered, async), each wave owns 16 rows.
// mode 0: bf16 out in head-split layout [B,H,S,64]   (Q and K)
// mode 1: bf16 out in transposed layout [B,H,64,S]   (V^T)
// mode 2: fp32 out row-major [4096,1024] + bias      (final projection)
__global__ void __launch_bounds__(256)
gemm_xwT_kernel(const __bf16* __restrict__ A, const __bf16* __restrict__ Wt,
                __bf16* __restrict__ out_bf, float* __restrict__ out_f,
                const float* __restrict__ bias, int mode) {
  __shared__ __bf16 ldsW[2][64 * KCH];            // 2 x 8 KiB

  const int tid  = threadIdx.x;
  const int lane = tid & 31;
  const int warp = tid >> 5;
  const int tm   = blockIdx.x >> 4;               // 0..31 (128-row tiles)
  const int tn   = blockIdx.x & 15;               // 0..15 (64-col strips)
  const int m0   = tm * 128 + warp * 16;
  const int n0   = tn * 64;
  const int lo   = lane & 15, half = lane >> 4;

  // Cooperative staging: thread t copies rows sr and sr+32, 8 bf16 each.
  const int sr = tid >> 3;                        // 0..31
  const int sc = (tid & 7) * 8;                   // 0,8,..,56
  const __bf16* wsrc = Wt + (size_t)(n0 + sr) * kD + sc;

  auto stage = [&](int buf, int kbase) {
    copy16_g2l(wsrc + kbase,                   &ldsW[buf][sr * KCH + sc]);
    copy16_g2l(wsrc + kbase + (size_t)32 * kD, &ldsW[buf][(sr + 32) * KCH + sc]);
  };

  v8f c0 = {}, c1 = {}, c2 = {}, c3 = {};
  const __bf16* arow = A + (size_t)(m0 + lo) * kD;

  stage(0, 0);
  for (int kk = 0; kk < kD / KCH; ++kk) {         // 16 chunks
    wait_async_copies();                          // own copies for buf kk&1 done
    __syncthreads();                              // everyone's copies visible;
                                                  // prev buffer's readers retired
    if (kk + 1 < kD / KCH) stage((kk + 1) & 1, (kk + 1) * KCH);

    const __bf16* wb = &ldsW[kk & 1][0];
#pragma unroll
    for (int ks = 0; ks < KCH; ks += 32) {
      v16bf a = ldfrag(arow, kk * KCH + ks, half);
      c0 = wmma_bf16(a, ldfrag(wb + ( 0 + lo) * KCH, ks, half), c0);
      c1 = wmma_bf16(a, ldfrag(wb + (16 + lo) * KCH, ks, half), c1);
      c2 = wmma_bf16(a, ldfrag(wb + (32 + lo) * KCH, ks, half), c2);
      c3 = wmma_bf16(a, ldfrag(wb + (48 + lo) * KCH, ks, half), c3);
    }
  }

  v8f cs[4] = {c0, c1, c2, c3};
#pragma unroll
  for (int f = 0; f < 4; ++f) {
#pragma unroll
    for (int r = 0; r < 8; ++r) {
      const int m = m0 + r + 8 * half;            // global row 0..4095
      const int n = n0 + f * 16 + lo;             // global col 0..1023
      const float v = cs[f][r];
      if (mode == 2) {
        out_f[(size_t)m * kD + n] = v + bias[n];
      } else {
        const int b = m >> 11, s = m & (kS - 1);
        const int h = n >> 6,  d = n & 63;
        if (mode == 0)
          out_bf[(((size_t)(b * kH + h)) * kS + s) * kHD + d] = (__bf16)v;
        else
          out_bf[(((size_t)(b * kH + h)) * kHD + d) * kS + s] = (__bf16)v;
      }
    }
  }
}

// ---------------------------------------------------------------------------
// Sliding-window attention. One wave per (b, h, 16-query-row tile).
// Online (flash) softmax over key-tile pairs; P transposed C->A layout via a
// 1KB per-wave LDS buffer (per-wave s_wait_dscnt; no cross-wave barrier,
// since waves in a block have different trip counts).
__global__ void __launch_bounds__(256)
attn_kernel(const __bf16* __restrict__ Q, const __bf16* __restrict__ K,
            const __bf16* __restrict__ Vt, __bf16* __restrict__ att) {
  __shared__ __bf16 lds_p[8][16 * 32];            // 8 waves x (16 rows x 32 j)

  const int lane = threadIdx.x & 31;
  const int warp = threadIdx.x >> 5;
  const int gid  = blockIdx.x * 8 + warp;         // 0 .. 4095
  const int qt   = gid & 127;                     // query tile in sequence
  const int bh   = gid >> 7;                      // b*16 + h, 0..31
  const int q0   = qt * 16;
  const int lo   = lane & 15, half = lane >> 4;

  const __bf16* qb = Q  + (size_t)bh * kS * kHD;
  const __bf16* kb = K  + (size_t)bh * kS * kHD;
  const __bf16* vb = Vt + (size_t)bh * kHD * kS;

  // Q fragments for this tile (16 rows x 64 K), reused for every key tile.
  const __bf16* qrow = qb + (size_t)(q0 + lo) * kHD;
  const v16bf aq0 = ldfrag(qrow, 0, half);
  const v16bf aq1 = ldfrag(qrow, 32, half);

  float run_max[8], run_sum[8];
  v8f o0 = {}, o1 = {}, o2 = {}, o3 = {};
#pragma unroll
  for (int r = 0; r < 8; ++r) { run_max[r] = -1e30f; run_sum[r] = 0.f; }

  const int jt_lo = (qt >= 16) ? (qt - 16) : 0;   // window covers <=17 tiles
  __bf16* lp = &lds_p[warp][0];

  for (int jt = jt_lo; jt <= qt; jt += 2) {
    // ---- scores for the tile pair: S = (Q K^T) / 8 -----------------------
    v8f s[2];
#pragma unroll
    for (int t = 0; t < 2; ++t) {
      int jr = (jt + t) * 16 + lo;
      if (jr > kS - 1) jr = kS - 1;               // clamp; masked below
      const __bf16* krow = kb + (size_t)jr * kHD;
      v8f c = {};
      c = wmma_bf16(aq0, ldfrag(krow, 0, half), c);
      c = wmma_bf16(aq1, ldfrag(krow, 32, half), c);
      s[t] = c;
    }

    // ---- mask + per-row max (reduce across the 16 lanes of a half) -------
    float tmax[8];
#pragma unroll
    for (int r = 0; r < 8; ++r) {
      const int i = q0 + r + 8 * half;
#pragma unroll
      for (int t = 0; t < 2; ++t) {
        const int j = (jt + t) * 16 + lo;
        const bool valid = (j <= i) && (j + kW > i);
        s[t][r] = valid ? s[t][r] * 0.125f : -1e30f;
      }
      float m2 = fmaxf(s[0][r], s[1][r]);
#pragma unroll
      for (int off = 1; off < 16; off <<= 1)
        m2 = fmaxf(m2, __shfl_xor(m2, off, 32));
      tmax[r] = m2;
    }

    // ---- online softmax update, write bf16 P into LDS --------------------
#pragma unroll
    for (int r = 0; r < 8; ++r) {
      const float nm   = fmaxf(run_max[r], tmax[r]);
      const float corr = __expf(run_max[r] - nm); // <= 1, ->0 when leaving -1e30
      run_max[r] = nm;
      run_sum[r] *= corr;
      o0[r] *= corr; o1[r] *= corr; o2[r] *= corr; o3[r] *= corr;

      const int i = q0 + r + 8 * half;
      float psum = 0.f;
#pragma unroll
      for (int t = 0; t < 2; ++t) {
        const int j = (jt + t) * 16 + lo;
        const bool valid = (j <= i) && (j + kW > i);
        const float p = valid ? __expf(s[t][r] - nm) : 0.f;
        psum += p;
        lp[(r + 8 * half) * 32 + t * 16 + lo] = (__bf16)p;
      }
#pragma unroll
      for (int off = 1; off < 16; off <<= 1)
        psum += __shfl_xor(psum, off, 32);
      run_sum[r] += psum;
    }

    // Per-wave DS drain before re-reading own LDS in transposed order.
    asm volatile("s_wait_dscnt 0x0" ::: "memory");

    const v16bf ap = ldfrag(lp + lo * 32, 0, half);

    // ---- O += P * V : V^T rows are contiguous in j -----------------------
    const int j0p = jt * 16;
    const int ka  = j0p + half * 8;
    int kbv = j0p + 16 + half * 8;
    if (kbv > kS - 8) kbv = kS - 8;               // clamp; P==0 there
#pragma unroll
    for (int f = 0; f < 4; ++f) {
      const __bf16* vrow = vb + (size_t)(f * 16 + lo) * kS;
      union { v16bf v; v8bf h[2]; } u;
      u.h[0] = *reinterpret_cast<const v8bf*>(vrow + ka);
      u.h[1] = *reinterpret_cast<const v8bf*>(vrow + kbv);
      v8f& o = (f == 0) ? o0 : (f == 1) ? o1 : (f == 2) ? o2 : o3;
      o = wmma_bf16(ap, u.v, o);
    }
  }

  // ---- normalize and store merged-head output (bf16, [B*S, 1024]) --------
#pragma unroll
  for (int r = 0; r < 8; ++r) {
    const float inv = 1.0f / run_sum[r];
    o0[r] *= inv; o1[r] *= inv; o2[r] *= inv; o3[r] *= inv;
  }
  const int bidx = bh >> 4, hidx = bh & 15;
  v8f os[4] = {o0, o1, o2, o3};
#pragma unroll
  for (int f = 0; f < 4; ++f) {
#pragma unroll
    for (int r = 0; r < 8; ++r) {
      const int m = q0 + r + 8 * half;
      const int d = hidx * kHD + f * 16 + lo;
      att[(size_t)(bidx * kS + m) * kD + d] = (__bf16)os[f][r];
    }
  }
}

// ---------------------------------------------------------------------------
extern "C" void kernel_launch(void* const* d_in, const int* in_sizes, int n_in,
                              void* d_out, int out_size, void* d_ws, size_t ws_size,
                              hipStream_t stream) {
  (void)in_sizes; (void)n_in; (void)out_size; (void)ws_size;
  const float* hs = (const float*)d_in[0];
  const float* Wq = (const float*)d_in[1];
  const float* Wk = (const float*)d_in[2];
  const float* Wv = (const float*)d_in[3];
  const float* Wo = (const float*)d_in[4];
  const float* bo = (const float*)d_in[5];
  float* out = (float*)d_out;

  char* ws = (char*)d_ws;                            // needs 48 MiB
  __bf16* xb   = (__bf16*)(ws);                      // [4096,1024]     8 MiB
  __bf16* wqb  = (__bf16*)(ws + (size_t)( 8 << 20)); // [1024,1024]     2 MiB
  __bf16* wkb  = (__bf16*)(ws + (size_t)(10 << 20));
  __bf16* wvb  = (__bf16*)(ws + (size_t)(12 << 20));
  __bf16* wob  = (__bf16*)(ws + (size_t)(14 << 20));
  __bf16* qb   = (__bf16*)(ws + (size_t)(16 << 20)); // [B,H,S,64]      8 MiB
  __bf16* kbuf = (__bf16*)(ws + (size_t)(24 << 20)); // [B,H,S,64]      8 MiB
  __bf16* vtb  = (__bf16*)(ws + (size_t)(32 << 20)); // [B,H,64,S]      8 MiB
  __bf16* attb = (__bf16*)(ws + (size_t)(40 << 20)); // [4096,1024]     8 MiB

  const int nX4 = kM * kD / 4, nW4 = kD * kD / 4;
  f32_to_bf16_kernel<<<(nX4 + 255) / 256, 256, 0, stream>>>(hs, xb,  nX4);
  f32_to_bf16_kernel<<<(nW4 + 255) / 256, 256, 0, stream>>>(Wq, wqb, nW4);
  f32_to_bf16_kernel<<<(nW4 + 255) / 256, 256, 0, stream>>>(Wk, wkb, nW4);
  f32_to_bf16_kernel<<<(nW4 + 255) / 256, 256, 0, stream>>>(Wv, wvb, nW4);
  f32_to_bf16_kernel<<<(nW4 + 255) / 256, 256, 0, stream>>>(Wo, wob, nW4);

  const int gblocks = (kM / 128) * (kD / 64);        // 512 blocks x 8 waves
  gemm_xwT_kernel<<<gblocks, 256, 0, stream>>>(xb, wqb, qb,   nullptr, nullptr, 0);
  gemm_xwT_kernel<<<gblocks, 256, 0, stream>>>(xb, wkb, kbuf, nullptr, nullptr, 0);
  gemm_xwT_kernel<<<gblocks, 256, 0, stream>>>(xb, wvb, vtb,  nullptr, nullptr, 1);

  const int atiles = kB * kH * (kS / 16);            // 4096 wave-tiles
  attn_kernel<<<atiles / 8, 256, 0, stream>>>(qb, kbuf, vtb, attb);

  gemm_xwT_kernel<<<gblocks, 256, 0, stream>>>(attb, wob, nullptr, out, bo, 2);
}
